// FPModule_12060268167710
// MI455X (gfx1250) — matrix operand order repaired and checked by reading, not compiled, not run on hardware
//
#include <hip/hip_runtime.h>

// ---------------------------------------------------------------------------
// Problem constants (match reference)
// ---------------------------------------------------------------------------
constexpr int B_     = 16;
constexpr int N_PER  = 1024;
constexpr int M_PER  = 4096;
constexpr int KNN    = 3;
constexpr int NVEC   = 64;
constexpr int F      = 3 * NVEC;   // 192
constexpr int F_SKIP = 64;
constexpr int HID    = 256;        // F + F_SKIP == HID
constexpr int OUTF   = 128;
constexpr int N_     = B_ * N_PER; // 16384
constexpr int M_     = B_ * M_PER; // 65536

typedef float    v8f  __attribute__((ext_vector_type(8)));
typedef _Float16 v16h __attribute__((ext_vector_type(16)));
typedef _Float16 v8h  __attribute__((ext_vector_type(8)));

// ---------------------------------------------------------------------------
// Workspace layout (bytes)
// ---------------------------------------------------------------------------
constexpr size_t WS_XIDX = 0;                                   // M*3 int32
constexpr size_t WS_PREP = WS_XIDX + (size_t)M_ * 3 * 4;        // M*32 f32 (3x9 Uw, padded)
constexpr size_t WS_HBUF = WS_PREP + (size_t)M_ * 32 * 4;       // M*256 f16
constexpr size_t WS_ABUF = WS_HBUF + (size_t)M_ * HID * 2;      // M*256 f16
constexpr size_t WS_W1T  = WS_ABUF + (size_t)M_ * HID * 2;      // 256*256 f16 (transposed)
constexpr size_t WS_W2T  = WS_W1T + (size_t)HID * HID * 2;      // 128*256 f16 (transposed)

// ---------------------------------------------------------------------------
// 1) Brute-force KNN: each thread owns one query, coarse points tiled in LDS
// ---------------------------------------------------------------------------
__global__ __launch_bounds__(256) void knn_kernel(
    const float* __restrict__ pos, const float* __restrict__ pos_skip,
    int* __restrict__ x_idx)
{
  const int tid = threadIdx.x;
  const int m   = blockIdx.x * 256 + tid;       // block never straddles a batch
  const int b   = m >> 12;                       // m / M_PER
  const int nb  = b * N_PER;

  const float qx = pos_skip[m * 3 + 0];
  const float qy = pos_skip[m * 3 + 1];
  const float qz = pos_skip[m * 3 + 2];

  float d0 = 3.0e38f, d1 = 3.0e38f, d2 = 3.0e38f;
  int   i0 = 0, i1 = 0, i2 = 0;

  __shared__ float sp[256 * 3];
  for (int t = 0; t < N_PER / 256; ++t) {
    __syncthreads();
    const int nl = t * 256 + tid;
    sp[tid * 3 + 0] = pos[(nb + nl) * 3 + 0];
    sp[tid * 3 + 1] = pos[(nb + nl) * 3 + 1];
    sp[tid * 3 + 2] = pos[(nb + nl) * 3 + 2];
    __syncthreads();
#pragma unroll 4
    for (int j = 0; j < 256; ++j) {
      const float dx = sp[j * 3 + 0] - qx;
      const float dy = sp[j * 3 + 1] - qy;
      const float dz = sp[j * 3 + 2] - qz;
      const float dd = dx * dx + dy * dy + dz * dz;
      const int   n  = nb + t * 256 + j;
      if (dd < d0)      { d2 = d1; i2 = i1; d1 = d0; i1 = i0; d0 = dd; i0 = n; }
      else if (dd < d1) { d2 = d1; i2 = i1; d1 = dd; i1 = n; }
      else if (dd < d2) { d2 = dd; i2 = n; }
    }
  }
  x_idx[m * 3 + 0] = i0;
  x_idx[m * 3 + 1] = i1;
  x_idx[m * 3 + 2] = i2;
}

// ---------------------------------------------------------------------------
// 2) Edge prep: per target node, fold normalized weights into U = Ry * Rx^T
//    prep[m][k*9 + a*3 + c] = (w_k / sum w) * sum_b Ry[a][b] * Rx_k[c][b]
// ---------------------------------------------------------------------------
__global__ __launch_bounds__(256) void prep_kernel(
    const float* __restrict__ pos, const float* __restrict__ pos_skip,
    const float* __restrict__ lframes, const float* __restrict__ lframes_skip,
    const int* __restrict__ x_idx, float* __restrict__ prep)
{
  const int m = blockIdx.x * 256 + threadIdx.x;

  const float qx = pos_skip[m * 3 + 0];
  const float qy = pos_skip[m * 3 + 1];
  const float qz = pos_skip[m * 3 + 2];

  int   idx[KNN];
  float w[KNN];
  float den = 0.0f;
#pragma unroll
  for (int k = 0; k < KNN; ++k) {
    idx[k] = x_idx[m * 3 + k];
    const float dx = pos[idx[k] * 3 + 0] - qx;
    const float dy = pos[idx[k] * 3 + 1] - qy;
    const float dz = pos[idx[k] * 3 + 2] - qz;
    const float sq = dx * dx + dy * dy + dz * dz;
    w[k] = 1.0f / fmaxf(sq, 1e-16f);
    den += w[k];
  }
  const float rden = 1.0f / den;

  float Ry[9];
#pragma unroll
  for (int j = 0; j < 9; ++j) Ry[j] = lframes_skip[m * 9 + j];

  float* pr = prep + (size_t)m * 32;
#pragma unroll
  for (int k = 0; k < KNN; ++k) {
    float Rx[9];
#pragma unroll
    for (int j = 0; j < 9; ++j) Rx[j] = lframes[idx[k] * 9 + j];
    const float s = w[k] * rden;
#pragma unroll
    for (int a = 0; a < 3; ++a)
#pragma unroll
      for (int c = 0; c < 3; ++c) {
        const float u = Ry[a * 3 + 0] * Rx[c * 3 + 0]
                      + Ry[a * 3 + 1] * Rx[c * 3 + 1]
                      + Ry[a * 3 + 2] * Rx[c * 3 + 2];
        pr[k * 9 + a * 3 + c] = s * u;
      }
  }
}

// ---------------------------------------------------------------------------
// 3) Transform + pack: thread (m, v) computes y[m, 3v..3v+2], writes f16 h row
//    h = [ y (192) | x_skip (64) ] in f16 for the WMMA MLP.  4 nodes / block.
// ---------------------------------------------------------------------------
__global__ __launch_bounds__(256) void transform_kernel(
    const float* __restrict__ x, const float* __restrict__ x_skip,
    const int* __restrict__ x_idx, const float* __restrict__ prep,
    _Float16* __restrict__ hbuf)
{
  const int tid = threadIdx.x;
  const int v   = tid & 63;
  const int ml  = tid >> 6;
  const int m   = blockIdx.x * 4 + ml;

  __shared__ float sp[4 * 32];
  if (tid < 128) sp[tid] = prep[(size_t)blockIdx.x * 4 * 32 + tid];
  __syncthreads();
  const float* Uw = sp + ml * 32;

  float y0 = 0.f, y1 = 0.f, y2 = 0.f;
#pragma unroll
  for (int k = 0; k < KNN; ++k) {
    const int ix = x_idx[m * 3 + k];
    const float* xv = x + (size_t)ix * F + v * 3;
    const float b0 = xv[0], b1 = xv[1], b2 = xv[2];
    const float* u = Uw + k * 9;
    y0 += u[0] * b0 + u[1] * b1 + u[2] * b2;
    y1 += u[3] * b0 + u[4] * b1 + u[5] * b2;
    y2 += u[6] * b0 + u[7] * b1 + u[8] * b2;
  }
  _Float16* hr = hbuf + (size_t)m * HID;
  hr[v * 3 + 0] = (_Float16)y0;
  hr[v * 3 + 1] = (_Float16)y1;
  hr[v * 3 + 2] = (_Float16)y2;
  hr[F + v]     = (_Float16)x_skip[(size_t)m * F_SKIP + v];
}

// ---------------------------------------------------------------------------
// 4) Convert weights to f16 transposed [N][K] (contiguous K for B-fragments)
// ---------------------------------------------------------------------------
__global__ __launch_bounds__(256) void weights_kernel(
    const float* __restrict__ W1, const float* __restrict__ W2,
    _Float16* __restrict__ w1t, _Float16* __restrict__ w2t)
{
  const int gid = blockIdx.x * 256 + threadIdx.x;
  if (gid < HID * HID) {
    const int n = gid >> 8, k = gid & 255;
    w1t[n * HID + k] = (_Float16)W1[(size_t)k * HID + n];
  } else {
    const int j = gid - HID * HID;           // 0 .. 128*256-1
    const int n = j >> 8, k = j & 255;
    w2t[n * HID + k] = (_Float16)W2[(size_t)k * OUTF + n];
  }
}

// ---------------------------------------------------------------------------
// 5) WMMA GEMM: C[M, LDC] = act(A[M,256] @ Bt^T + bias)
//    Block: 8 waves, 128x128 tile. Wave: 32x64 tile = 2x4 accum of 16x16.
//    A fragment per lane (l2,hi): row l2, K = hi*16..hi*16+15
//    B fragment per lane (l2,hi): col l2, K = hi*16..hi*16+15 (Bt row-major)
//    D lane (l2,hi): VGPR j -> C[hi*8+j][l2]
// ---------------------------------------------------------------------------
template <int LDC, bool RELU, bool OUTF16>
__global__ __launch_bounds__(256) void gemm_wmma(
    const _Float16* __restrict__ A, const _Float16* __restrict__ Bt,
    const float* __restrict__ bias, float* __restrict__ Cf,
    _Float16* __restrict__ Ch)
{
  const int tid  = threadIdx.x;
  const int wv   = tid >> 5;
  const int lane = tid & 31;
  const int l2   = lane & 15;
  const int hi   = lane >> 4;
  const int wm   = wv & 3;              // 4 M strips of 32
  const int wn   = wv >> 2;             // 2 N strips of 64
  const int mbase = blockIdx.y * 128 + wm * 32;
  const int nbase = blockIdx.x * 128 + wn * 64;

  v8f acc[2][4] = {};

  for (int kk = 0; kk < HID; kk += 32) {
    v16h af[2];
#pragma unroll
    for (int ms = 0; ms < 2; ++ms) {
      const _Float16* pa =
          A + (size_t)(mbase + ms * 16 + l2) * HID + kk + hi * 16;
      const v8h lo = *(const v8h*)pa;
      const v8h hh = *(const v8h*)(pa + 8);
#pragma unroll
      for (int e = 0; e < 8; ++e) { af[ms][e] = lo[e]; af[ms][8 + e] = hh[e]; }
    }
    v16h bf[4];
#pragma unroll
    for (int ns = 0; ns < 4; ++ns) {
      const _Float16* pb =
          Bt + (size_t)(nbase + ns * 16 + l2) * HID + kk + hi * 16;
      bf[ns] = *(const v16h*)pb;
    }
#pragma unroll
    for (int ms = 0; ms < 2; ++ms)
#pragma unroll
      for (int ns = 0; ns < 4; ++ns)
        acc[ms][ns] = __builtin_amdgcn_wmma_f32_16x16x32_f16(
            false, af[ms], false, bf[ns], (short)0, acc[ms][ns], false, false);
  }

#pragma unroll
  for (int ms = 0; ms < 2; ++ms)
#pragma unroll
    for (int ns = 0; ns < 4; ++ns) {
      const int   col  = nbase + ns * 16 + l2;
      const float bv   = bias[col];
      const int   row0 = mbase + ms * 16 + hi * 8;
#pragma unroll
      for (int j = 0; j < 8; ++j) {
        float vv = acc[ms][ns][j] + bv;
        if (RELU) vv = fmaxf(vv, 0.0f);
        if (OUTF16) Ch[(size_t)(row0 + j) * LDC + col] = (_Float16)vv;
        else        Cf[(size_t)(row0 + j) * LDC + col] = vv;
      }
    }
}

// ---------------------------------------------------------------------------
// 6) Tail: copy pos_skip, batch_skip (uniform pattern), lframes_skip
// ---------------------------------------------------------------------------
__global__ __launch_bounds__(256) void tail_kernel(
    const float* __restrict__ pos_skip, const float* __restrict__ lframes_skip,
    float* __restrict__ out)
{
  const int m = blockIdx.x * 256 + threadIdx.x;
  float* o_pos = out + (size_t)M_ * OUTF;
  float* o_bat = o_pos + (size_t)M_ * 3;
  float* o_lf  = o_bat + (size_t)M_;
#pragma unroll
  for (int i = 0; i < 3; ++i) o_pos[m * 3 + i] = pos_skip[m * 3 + i];
  o_bat[m] = (float)(m >> 12);   // m / M_PER
#pragma unroll
  for (int i = 0; i < 9; ++i) o_lf[m * 9 + i] = lframes_skip[m * 9 + i];
}

// ---------------------------------------------------------------------------
// Host launcher
// ---------------------------------------------------------------------------
extern "C" void kernel_launch(void* const* d_in, const int* in_sizes, int n_in,
                              void* d_out, int out_size, void* d_ws,
                              size_t ws_size, hipStream_t stream)
{
  const float* x            = (const float*)d_in[0];
  const float* pos          = (const float*)d_in[1];
  const float* pos_skip     = (const float*)d_in[2];
  const float* x_skip       = (const float*)d_in[3];
  const float* lframes      = (const float*)d_in[4];
  const float* lframes_skip = (const float*)d_in[5];
  const float* W1           = (const float*)d_in[6];
  const float* b1           = (const float*)d_in[7];
  const float* W2           = (const float*)d_in[8];
  const float* b2           = (const float*)d_in[9];
  float*       out          = (float*)d_out;

  char* ws = (char*)d_ws;
  int*       x_idx = (int*)(ws + WS_XIDX);
  float*     prep  = (float*)(ws + WS_PREP);
  _Float16*  hbuf  = (_Float16*)(ws + WS_HBUF);
  _Float16*  abuf  = (_Float16*)(ws + WS_ABUF);
  _Float16*  w1t   = (_Float16*)(ws + WS_W1T);
  _Float16*  w2t   = (_Float16*)(ws + WS_W2T);

  knn_kernel<<<M_ / 256, 256, 0, stream>>>(pos, pos_skip, x_idx);

  prep_kernel<<<M_ / 256, 256, 0, stream>>>(pos, pos_skip, lframes,
                                            lframes_skip, x_idx, prep);

  transform_kernel<<<M_ / 4, 256, 0, stream>>>(x, x_skip, x_idx, prep, hbuf);

  weights_kernel<<<(HID * HID + OUTF * HID) / 256, 256, 0, stream>>>(W1, W2,
                                                                     w1t, w2t);

  // GEMM1: [M,256] x [256,256] + b1, ReLU -> f16
  gemm_wmma<HID, true, true><<<dim3(HID / 128, M_ / 128), 256, 0, stream>>>(
      hbuf, w1t, b1, nullptr, abuf);

  // GEMM2: [M,256] x [256,128] + b2 -> f32 out
  gemm_wmma<OUTF, false, false><<<dim3(OUTF / 128, M_ / 128), 256, 0, stream>>>(
      abuf, w2t, b2, out, nullptr);

  tail_kernel<<<M_ / 256, 256, 0, stream>>>(pos_skip, lframes_skip, out);
}
